// CausalMultiHeadAttention_4415226380965
// MI455X (gfx1250) — matrix operand order repaired and checked by reading, not compiled, not run on hardware
//
#include <hip/hip_runtime.h>

#define DI __device__ __forceinline__

typedef __attribute__((ext_vector_type(16))) __bf16 v16bf;
typedef __attribute__((ext_vector_type(8)))  float  v8f;

union Frag { unsigned int u[8]; v16bf v; };

DI v8f vzero() {
    v8f z;
#pragma unroll
    for (int i = 0; i < 8; ++i) z[i] = 0.0f;
    return z;
}

// fp32 -> bf16 via native conversion (RTNE); backend selects v_cvt_* if present
DI unsigned short bf16bits(float f) {
    union { __bf16 h; unsigned short u; } cv;
    cv.h = (__bf16)f;
    return cv.u;
}

#if __has_builtin(__builtin_amdgcn_cvt_pk_bf16_f32)
typedef __attribute__((ext_vector_type(2))) __bf16 v2bf;
DI unsigned int pack_bf16x2(float lo, float hi) {
    union { v2bf v; unsigned int u; } cv;
    cv.v = __builtin_amdgcn_cvt_pk_bf16_f32(lo, hi);
    return cv.u;
}
#else
DI unsigned int pack_bf16x2(float lo, float hi) {
    union { __bf16 h[2]; unsigned int u; } cv;
    cv.h[0] = (__bf16)lo;
    cv.h[1] = (__bf16)hi;
    return cv.u;
}
#endif

// A-fragment K index (16-bit A 16x32): r<4 -> 2r else 16+2(r-4); +8*half
DI int kbA(int r, int half) { return ((r < 4) ? 2 * r : 16 + 2 * (r - 4)) + 8 * half; }
// B-fragment K index (16-bit B 32x16): r<4 -> 2r else 16+2(r-4); +8*half
DI int kbB(int r, int half) { return ((r < 4) ? 2 * r : 16 + 2 * (r - 4)) + 8 * half; }

DI v8f wmma_bf16(const Frag& a, const Frag& b, v8f c) {
    return __builtin_amdgcn_wmma_f32_16x16x32_bf16(false, a.v, false, b.v,
                                                   (short)0, c, false, false);
}

// ---- CDNA5 async global->LDS (ASYNCcnt-tracked, GV mode) ----
// the instruction offset is added to both the LDS address and the global address
DI unsigned lds_off(const void* p) { return (unsigned)(unsigned long long)p; }

DI void async_lds_b128x2(unsigned lds, const void* g) {
    asm volatile("global_load_async_to_lds_b128 %0, %1, off offset:0"
                 :: "v"(lds), "v"(g) : "memory");
    asm volatile("global_load_async_to_lds_b128 %0, %1, off offset:16"
                 :: "v"(lds), "v"(g) : "memory");
}
DI void async_lds_b128x4(unsigned lds, const void* g) {
    asm volatile("global_load_async_to_lds_b128 %0, %1, off offset:0"
                 :: "v"(lds), "v"(g) : "memory");
    asm volatile("global_load_async_to_lds_b128 %0, %1, off offset:16"
                 :: "v"(lds), "v"(g) : "memory");
    asm volatile("global_load_async_to_lds_b128 %0, %1, off offset:32"
                 :: "v"(lds), "v"(g) : "memory");
    asm volatile("global_load_async_to_lds_b128 %0, %1, off offset:48"
                 :: "v"(lds), "v"(g) : "memory");
}
DI void wait_async0() {
#if __has_builtin(__builtin_amdgcn_s_wait_asynccnt)
    __builtin_amdgcn_s_wait_asynccnt(0);
#else
    asm volatile("s_wait_asynccnt 0x0" ::: "memory");
#endif
}

// ---------------------------------------------------------------------------
// QKV GEMM: out[m,n] = sum_k X[m,k]*W[n,k] + bias[n]; bf16 out in [B,H,S,HD].
// 128x128 block tile, 8 waves of 32x64, K-step 32, double-buffered LDS.
// ---------------------------------------------------------------------------
__global__ __launch_bounds__(256)
void gemm_qkv_kernel(const float* __restrict__ X, const float* __restrict__ W,
                     const float* __restrict__ bias, unsigned short* __restrict__ out) {
    __shared__ __align__(16) unsigned short As[2][128 * 40];
    __shared__ __align__(16) unsigned short Bs[2][128 * 40];

    const int tid  = threadIdx.x;
    const int lane = tid & 31, wid = tid >> 5;
    const int lm = lane & 15, half = lane >> 4;
    const int waveM = (wid & 3) * 32;
    const int waveN = (wid >> 2) * 64;
    const int blkM = blockIdx.y * 128;
    const int blkN = blockIdx.x * 128;

    v8f acc[2][4];
#pragma unroll
    for (int i = 0; i < 2; ++i)
#pragma unroll
        for (int j = 0; j < 4; ++j) acc[i][j] = vzero();

    const int row  = tid >> 1;
    const int csel = (tid & 1) * 16;

    uint4 pa0, pa1, pb0, pb1;
    auto load_cvt = [&](int k0) {
        const float4* ag = (const float4*)(X + (size_t)(blkM + row) * 2048 + k0 + csel);
        const float4* bg = (const float4*)(W + (size_t)(blkN + row) * 2048 + k0 + csel);
        float4 a0 = ag[0], a1 = ag[1], a2 = ag[2], a3 = ag[3];
        float4 b0 = bg[0], b1 = bg[1], b2 = bg[2], b3 = bg[3];
        pa0 = {pack_bf16x2(a0.x, a0.y), pack_bf16x2(a0.z, a0.w),
               pack_bf16x2(a1.x, a1.y), pack_bf16x2(a1.z, a1.w)};
        pa1 = {pack_bf16x2(a2.x, a2.y), pack_bf16x2(a2.z, a2.w),
               pack_bf16x2(a3.x, a3.y), pack_bf16x2(a3.z, a3.w)};
        pb0 = {pack_bf16x2(b0.x, b0.y), pack_bf16x2(b0.z, b0.w),
               pack_bf16x2(b1.x, b1.y), pack_bf16x2(b1.z, b1.w)};
        pb1 = {pack_bf16x2(b2.x, b2.y), pack_bf16x2(b2.z, b2.w),
               pack_bf16x2(b3.x, b3.y), pack_bf16x2(b3.z, b3.w)};
    };
    auto store_tile = [&](int buf) {
        *(uint4*)&As[buf][row * 40 + csel]     = pa0;
        *(uint4*)&As[buf][row * 40 + csel + 8] = pa1;
        *(uint4*)&Bs[buf][row * 40 + csel]     = pb0;
        *(uint4*)&Bs[buf][row * 40 + csel + 8] = pb1;
    };
    auto compute = [&](int buf) {
        Frag af[2], bf[4];
#pragma unroll
        for (int mt = 0; mt < 2; ++mt) {
            const unsigned short* base = &As[buf][(waveM + mt * 16 + lm) * 40];
#pragma unroll
            for (int r = 0; r < 8; ++r)
                af[mt].u[r] = *(const unsigned int*)&base[kbA(r, half)];
        }
#pragma unroll
        for (int nt = 0; nt < 4; ++nt) {
            const unsigned short* base = &Bs[buf][(waveN + nt * 16 + lm) * 40];
#pragma unroll
            for (int r = 0; r < 8; ++r)
                bf[nt].u[r] = *(const unsigned int*)&base[kbB(r, half)];
        }
#pragma unroll
        for (int mt = 0; mt < 2; ++mt)
#pragma unroll
            for (int nt = 0; nt < 4; ++nt)
                acc[mt][nt] = wmma_bf16(af[mt], bf[nt], acc[mt][nt]);
    };

    load_cvt(0);
    store_tile(0);
    __syncthreads();
    for (int t = 0; t < 63; ++t) {
        const int cur = t & 1;
        load_cvt(32 * (t + 1));
        compute(cur);
        store_tile(1 - cur);
        __syncthreads();
    }
    compute(1);

#pragma unroll
    for (int mt = 0; mt < 2; ++mt)
#pragma unroll
        for (int nt = 0; nt < 4; ++nt) {
            const int n  = blkN + waveN + nt * 16 + lm;
            const float bn = bias[n];
            const int h = n >> 7, hd = n & 127;
#pragma unroll
            for (int r = 0; r < 8; ++r) {
                const int m = blkM + waveM + mt * 16 + half * 8 + r;
                const int b = m >> 11, s = m & 2047;
                const float v = acc[mt][nt][r] + bn;
                out[((size_t)(b * 16 + h) * 2048 + s) * 128 + hd] = bf16bits(v);
            }
        }
}

// ---------------------------------------------------------------------------
// Output projection: fp32 out[m,n] = sum_k CTX_bf16[m,k]*W[n,k] + bias[n].
// A-tile staged with async global->LDS; W converted in-register. Double-buffered.
// ---------------------------------------------------------------------------
__global__ __launch_bounds__(256)
void gemm_out_kernel(const unsigned short* __restrict__ Xb, const float* __restrict__ W,
                     const float* __restrict__ bias, float* __restrict__ out) {
    __shared__ __align__(16) unsigned short As[2][128 * 40];
    __shared__ __align__(16) unsigned short Bs[2][128 * 40];

    const int tid  = threadIdx.x;
    const int lane = tid & 31, wid = tid >> 5;
    const int lm = lane & 15, half = lane >> 4;
    const int waveM = (wid & 3) * 32;
    const int waveN = (wid >> 2) * 64;
    const int blkM = blockIdx.y * 128;
    const int blkN = blockIdx.x * 128;

    v8f acc[2][4];
#pragma unroll
    for (int i = 0; i < 2; ++i)
#pragma unroll
        for (int j = 0; j < 4; ++j) acc[i][j] = vzero();

    const int row  = tid >> 1;
    const int csel = (tid & 1) * 16;

    uint4 pb0, pb1;
    auto issue_asyncA = [&](int k0, int buf) {
        const unsigned la = lds_off(&As[buf][row * 40 + csel]);
        const unsigned short* ga = Xb + (size_t)(blkM + row) * 2048 + k0 + csel;
        async_lds_b128x2(la, ga);  // 32 bytes = 16 bf16
    };
    auto loadB_cvt = [&](int k0) {
        const float4* bg = (const float4*)(W + (size_t)(blkN + row) * 2048 + k0 + csel);
        float4 b0 = bg[0], b1 = bg[1], b2 = bg[2], b3 = bg[3];
        pb0 = {pack_bf16x2(b0.x, b0.y), pack_bf16x2(b0.z, b0.w),
               pack_bf16x2(b1.x, b1.y), pack_bf16x2(b1.z, b1.w)};
        pb1 = {pack_bf16x2(b2.x, b2.y), pack_bf16x2(b2.z, b2.w),
               pack_bf16x2(b3.x, b3.y), pack_bf16x2(b3.z, b3.w)};
    };
    auto storeB = [&](int buf) {
        *(uint4*)&Bs[buf][row * 40 + csel]     = pb0;
        *(uint4*)&Bs[buf][row * 40 + csel + 8] = pb1;
    };
    auto compute = [&](int buf) {
        Frag af[2], bf[4];
#pragma unroll
        for (int mt = 0; mt < 2; ++mt) {
            const unsigned short* base = &As[buf][(waveM + mt * 16 + lm) * 40];
#pragma unroll
            for (int r = 0; r < 8; ++r)
                af[mt].u[r] = *(const unsigned int*)&base[kbA(r, half)];
        }
#pragma unroll
        for (int nt = 0; nt < 4; ++nt) {
            const unsigned short* base = &Bs[buf][(waveN + nt * 16 + lm) * 40];
#pragma unroll
            for (int r = 0; r < 8; ++r)
                bf[nt].u[r] = *(const unsigned int*)&base[kbB(r, half)];
        }
#pragma unroll
        for (int mt = 0; mt < 2; ++mt)
#pragma unroll
            for (int nt = 0; nt < 4; ++nt)
                acc[mt][nt] = wmma_bf16(af[mt], bf[nt], acc[mt][nt]);
    };

    issue_asyncA(0, 0);
    loadB_cvt(0);
    storeB(0);
    wait_async0();
    __syncthreads();
    for (int t = 0; t < 63; ++t) {
        const int cur = t & 1;
        issue_asyncA(32 * (t + 1), 1 - cur);
        loadB_cvt(32 * (t + 1));
        compute(cur);
        storeB(1 - cur);
        wait_async0();
        __syncthreads();
    }
    compute(1);

#pragma unroll
    for (int mt = 0; mt < 2; ++mt)
#pragma unroll
        for (int nt = 0; nt < 4; ++nt) {
            const int n  = blkN + waveN + nt * 16 + lm;
            const float bn = bias[n];
#pragma unroll
            for (int r = 0; r < 8; ++r) {
                const int m = blkM + waveM + mt * 16 + half * 8 + r;
                out[(size_t)m * 2048 + n] = acc[mt][nt][r] + bn;
            }
        }
}

// ---------------------------------------------------------------------------
// Causal flash attention. Block = (b*h, 128 q rows); 8 waves x 16 q rows.
// K-tile staged via async global->LDS; V transposed manually; P via LDS relayout.
// ---------------------------------------------------------------------------
__global__ __launch_bounds__(256)
void attn_kernel(const unsigned short* __restrict__ Q, const unsigned short* __restrict__ Kw,
                 const unsigned short* __restrict__ Vw, unsigned short* __restrict__ CTX) {
    __shared__ __align__(16) unsigned short Ks[64 * 136];   // [j][d]
    __shared__ __align__(16) unsigned short Vt[128 * 72];   // [d][j]
    __shared__ __align__(16) unsigned short Ps[128 * 72];   // per-wave 16 rows

    const int bh = blockIdx.x, qb = blockIdx.y;
    const int b = bh >> 4, h = bh & 15;
    const int tid = threadIdx.x, lane = tid & 31, wid = tid >> 5;
    const int lm = lane & 15, half = lane >> 4;
    const size_t head_off = (size_t)bh * 2048 * 128;

    Frag qf[4];
    {
        const int qrow_l = qb * 128 + wid * 16 + lm;
        const unsigned short* Qg = Q + head_off + (size_t)qrow_l * 128;
#pragma unroll
        for (int kk = 0; kk < 4; ++kk)
#pragma unroll
            for (int r = 0; r < 8; ++r)
                qf[kk].u[r] = *(const unsigned int*)&Qg[kk * 32 + kbA(r, half)];
    }

    v8f O[8];
    float mi[8], li[8];
#pragma unroll
    for (int dt = 0; dt < 8; ++dt) O[dt] = vzero();
#pragma unroll
    for (int r = 0; r < 8; ++r) { mi[r] = -1.0e30f; li[r] = 0.0f; }

    const float scale = 0.08838834764831845f;  // 1/sqrt(128)
    const int qrow_base = qb * 128 + wid * 16 + half * 8;
    const int nkt = (qb + 1) * 2;

    for (int kt = 0; kt < nkt; ++kt) {
        const int j0 = kt * 64;
        __syncthreads();  // previous iteration's LDS reads done
        {
            // K tile 64x128 bf16: async copy straight into LDS
            const int j = tid >> 2, dseg = (tid & 3) * 32;
            const unsigned lk = lds_off(&Ks[j * 136 + dseg]);
            const unsigned short* gk = Kw + head_off + (size_t)(j0 + j) * 128 + dseg;
            async_lds_b128x4(lk, gk);  // 64 bytes
        }
        {
            // V tile transposed into Vt[d][j]
            const int j = tid >> 2, dseg = (tid & 3) * 32;
            const unsigned int* src =
                (const unsigned int*)(Vw + head_off + (size_t)(j0 + j) * 128 + dseg);
#pragma unroll
            for (int i = 0; i < 16; ++i) {
                const unsigned int w = src[i];
                const int d = dseg + 2 * i;
                Vt[d * 72 + j]       = (unsigned short)(w & 0xFFFFu);
                Vt[(d + 1) * 72 + j] = (unsigned short)(w >> 16);
            }
        }
        wait_async0();
        __syncthreads();

        // S = Q K^T (4 column tiles of 16 keys)
        v8f sc[4];
#pragma unroll
        for (int nt = 0; nt < 4; ++nt) {
            sc[nt] = vzero();
#pragma unroll
            for (int kk = 0; kk < 4; ++kk) {
                Frag bfk;
                const unsigned short* base = &Ks[(nt * 16 + lm) * 136 + kk * 32];
#pragma unroll
                for (int r = 0; r < 8; ++r)
                    bfk.u[r] = *(const unsigned int*)&base[kbB(r, half)];
                sc[nt] = wmma_bf16(qf[kk], bfk, sc[nt]);
            }
        }

        // scale + causal mask
#pragma unroll
        for (int nt = 0; nt < 4; ++nt)
#pragma unroll
            for (int r = 0; r < 8; ++r) {
                float v = sc[nt][r] * scale;
                const int j = j0 + nt * 16 + lm;
                if (j > qrow_base + r) v = -1.0e30f;
                sc[nt][r] = v;
            }

        // online softmax (rows per reg; 16-lane shfl reductions stay within a half)
        float corr[8], rsum[8];
#pragma unroll
        for (int r = 0; r < 8; ++r) {
            float mx = fmaxf(fmaxf(sc[0][r], sc[1][r]), fmaxf(sc[2][r], sc[3][r]));
#pragma unroll
            for (int off = 1; off < 16; off <<= 1)
                mx = fmaxf(mx, __shfl_xor(mx, off, 32));
            const float mnew = fmaxf(mi[r], mx);
            corr[r] = __expf(mi[r] - mnew);
            float s = 0.0f;
#pragma unroll
            for (int nt = 0; nt < 4; ++nt) {
                const float p = __expf(sc[nt][r] - mnew);
                sc[nt][r] = p;
                s += p;
            }
#pragma unroll
            for (int off = 1; off < 16; off <<= 1)
                s += __shfl_xor(s, off, 32);
            rsum[r] = s;
            mi[r] = mnew;
        }
#pragma unroll
        for (int r = 0; r < 8; ++r) li[r] = li[r] * corr[r] + rsum[r];

        // rescale O; stash P bf16 to per-wave LDS (C-layout -> A-layout)
#pragma unroll
        for (int dt = 0; dt < 8; ++dt)
#pragma unroll
            for (int r = 0; r < 8; ++r) O[dt][r] *= corr[r];
#pragma unroll
        for (int nt = 0; nt < 4; ++nt)
#pragma unroll
            for (int r = 0; r < 8; ++r)
                Ps[(wid * 16 + half * 8 + r) * 72 + nt * 16 + lm] = bf16bits(sc[nt][r]);
        __syncthreads();

        // O += P V
#pragma unroll
        for (int kkj = 0; kkj < 2; ++kkj) {
            Frag pa;
            const unsigned short* pb = &Ps[(wid * 16 + lm) * 72 + kkj * 32];
#pragma unroll
            for (int r = 0; r < 8; ++r)
                pa.u[r] = *(const unsigned int*)&pb[kbA(r, half)];
#pragma unroll
            for (int dt = 0; dt < 8; ++dt) {
                Frag vb;
                const unsigned short* vbase = &Vt[(dt * 16 + lm) * 72 + kkj * 32];
#pragma unroll
                for (int r = 0; r < 8; ++r)
                    vb.u[r] = *(const unsigned int*)&vbase[kbB(r, half)];
                O[dt] = wmma_bf16(pa, vb, O[dt]);
            }
        }
    }

#pragma unroll
    for (int r = 0; r < 8; ++r) li[r] = 1.0f / li[r];
#pragma unroll
    for (int dt = 0; dt < 8; ++dt) {
        const int d = dt * 16 + lm;
#pragma unroll
        for (int r = 0; r < 8; ++r) {
            const float v = O[dt][r] * li[r];
            const int s = qrow_base + r;
            CTX[((size_t)(b * 2048 + s) * 2048) + h * 128 + d] = bf16bits(v);
        }
    }
}

// ---------------------------------------------------------------------------
extern "C" void kernel_launch(void* const* d_in, const int* in_sizes, int n_in,
                              void* d_out, int out_size, void* d_ws, size_t ws_size,
                              hipStream_t stream) {
    const float* hs = (const float*)d_in[0];
    // d_in[1] = attention_mask (tril bool) — causal mask applied analytically
    const float* Wq = (const float*)d_in[2];
    const float* bq = (const float*)d_in[3];
    const float* Wk = (const float*)d_in[4];
    const float* bk = (const float*)d_in[5];
    const float* Wv = (const float*)d_in[6];
    const float* bv = (const float*)d_in[7];
    const float* Wo = (const float*)d_in[8];
    const float* bo = (const float*)d_in[9];

    const size_t NE = (size_t)2 * 16 * 2048 * 128;  // 8,388,608 elems per tensor
    unsigned short* Qw = (unsigned short*)d_ws;
    unsigned short* Kw = Qw + NE;
    unsigned short* Vw = Kw + NE;
    unsigned short* Cw = Vw + NE;

    dim3 gg(16, 32), bb(256);
    gemm_qkv_kernel<<<gg, bb, 0, stream>>>(hs, Wq, bq, Qw);
    gemm_qkv_kernel<<<gg, bb, 0, stream>>>(hs, Wk, bk, Kw);
    gemm_qkv_kernel<<<gg, bb, 0, stream>>>(hs, Wv, bv, Vw);
    attn_kernel<<<dim3(32, 16), bb, 0, stream>>>(Qw, Kw, Vw, Cw);
    gemm_out_kernel<<<gg, bb, 0, stream>>>(Cw, Wo, bo, (float*)d_out);
}
